// NGCF_9268539425059
// MI455X (gfx1250) — matrix operand order repaired and checked by reading, not compiled, not run on hardware
//
#include <hip/hip_runtime.h>
#include <hip/hip_bf16.h>
#include <math.h>

#define N_USERS 100000
#define N_ITEMS 50000
#define N_NODES (N_USERS + N_ITEMS)
#define NNZ_EDGES 4800000
#define EMB 64
#define N_LAYERS 3
#define BATCH 4096
#define REG_COEF 1e-5f

typedef __attribute__((ext_vector_type(16))) _Float16 v16h;
typedef __attribute__((ext_vector_type(8)))  float    v8f;

// -------------------- concat user/item embeddings into x0 --------------------
__global__ void ngcf_concat(const float* __restrict__ ue, const float* __restrict__ ie,
                            float* __restrict__ x0) {
    size_t idx = (size_t)blockIdx.x * blockDim.x + threadIdx.x;
    size_t total = (size_t)N_NODES * EMB;
    if (idx >= total) return;
    size_t ubound = (size_t)N_USERS * EMB;
    x0[idx] = (idx < ubound) ? ue[idx] : ie[idx - ubound];
}

// -------------------- zero a float buffer --------------------
__global__ void ngcf_zero(float* __restrict__ p, size_t n) {
    size_t idx = (size_t)blockIdx.x * blockDim.x + threadIdx.x;
    if (idx < n) p[idx] = 0.0f;
}

// ---- one-shot: convert all layer weights fp32 -> f16 (B operands of WMMA) ---
__global__ void ngcf_cvt_weights(const float* __restrict__ W1, const float* __restrict__ W2,
                                 _Float16* __restrict__ W1h, _Float16* __restrict__ W2h) {
    int idx = blockIdx.x * blockDim.x + threadIdx.x;       // over 3*64*64 elements
    if (idx >= N_LAYERS * EMB * EMB) return;
    W1h[idx] = (_Float16)W1[idx];
    W2h[idx] = (_Float16)W2[idx];
}

// -------------------- SpMM: y[row] += val * x[col]  (one wave per edge) ------
__global__ void ngcf_spmm(const int* __restrict__ rows, const int* __restrict__ cols,
                          const float* __restrict__ vals, const float* __restrict__ x,
                          float* __restrict__ y) {
    const int lane   = threadIdx.x & 31;
    const int wave   = (int)((blockIdx.x * blockDim.x + threadIdx.x) >> 5);
    const int nwaves = (int)((gridDim.x * blockDim.x) >> 5);
    for (int e = wave; e < NNZ_EDGES; e += nwaves) {
        const int r = rows[e];                 // wave-uniform -> scalarized loads
        const int c = cols[e];
        const float v = vals[e];
        const float2 xe = *(const float2*)(x + (size_t)c * EMB + lane * 2);
        float* yp = y + (size_t)r * EMB + lane * 2;
        atomicAdd(yp,     v * xe.x);           // L2-resident atomic scatter
        atomicAdd(yp + 1, v * xe.y);
    }
}

// ---- fused: simple = (sL+x)@W1+b1 ; inter = (sL*x)@W2+b2 ;
// ----        out = rownorm(leaky_relu(simple+inter)); one wave = 16 rows -----
__global__ void ngcf_transform(const float* __restrict__ sL, const float* __restrict__ xin,
                               const _Float16* __restrict__ W1h, const float* __restrict__ b1,
                               const _Float16* __restrict__ W2h, const float* __restrict__ b2,
                               float* __restrict__ xout, int ntiles) {
    const int lane = threadIdx.x & 31;
    const int wave = (int)((blockIdx.x * blockDim.x + threadIdx.x) >> 5);
    if (wave >= ntiles) return;          // wave-uniform exit: EXEC stays all-1s for WMMA
    const int half = lane >> 4;          // 0: lanes 0-15, 1: lanes 16-31
    const int m    = lane & 15;
    const int row0 = wave * 16;

    // C/D layout (ISA 7.12.2): acc[nb][v] = out[M = v + 8*half][N = nb*16 + m]
    v8f acc[4];
    for (int nb = 0; nb < 4; ++nb) {
        const float bsum = b1[nb * 16 + m] + b2[nb * 16 + m];
        for (int v = 0; v < 8; ++v) acc[nb][v] = bsum;
    }

    for (int kb = 0; kb < 2; ++kb) {
        // A layout (16-bit A 16x32): lane holds row M=m; VGPR v packs K pair
        // K0 = kb*32 + (v>>2)*16 + half*8 + (v&3)*2
        v16h a1, a2;
        const float* sRow = sL  + (size_t)(row0 + m) * EMB;
        const float* xRow = xin + (size_t)(row0 + m) * EMB;
        for (int v = 0; v < 8; ++v) {
            const int K0 = kb * 32 + ((v >> 2) * 16) + half * 8 + ((v & 3) * 2);
            const float2 s = *(const float2*)(sRow + K0);
            const float2 e = *(const float2*)(xRow + K0);
            a1[2 * v]     = (_Float16)(s.x + e.x);   // (L+I)@x operand
            a1[2 * v + 1] = (_Float16)(s.y + e.y);
            a2[2 * v]     = (_Float16)(s.x * e.x);   // (L@x)*x operand
            a2[2 * v + 1] = (_Float16)(s.y * e.y);
        }
        // B layout (16-bit B 32x16): lane holds column N = nb*16 + m; element s has
        // K = kb*32 + (s>>3)*16 + half*8 + (s&7) (same K order as A).
        const int krow = kb * 32 + half * 8;
        for (int nb = 0; nb < 4; ++nb) {
            v16h bw1, bw2;
            for (int s = 0; s < 16; ++s) {
                const size_t o = (size_t)(krow + (s >> 3) * 16 + (s & 7)) * EMB + nb * 16 + m;
                bw1[s] = W1h[o];
                bw2[s] = W2h[o];
            }
            acc[nb] = __builtin_amdgcn_wmma_f32_16x16x32_f16(
                false, a1, false, bw1, (short)0, acc[nb], false, false);
            acc[nb] = __builtin_amdgcn_wmma_f32_16x16x32_f16(
                false, a2, false, bw2, (short)0, acc[nb], false, false);
        }
    }

    // leaky_relu + per-row L2 norm directly on WMMA accumulator layout
    float val[4][8];
    float ss[8];
    for (int v = 0; v < 8; ++v) ss[v] = 0.0f;
    for (int nb = 0; nb < 4; ++nb)
        for (int v = 0; v < 8; ++v) {
            float t = acc[nb][v];
            t = (t > 0.0f) ? t : 0.01f * t;
            val[nb][v] = t;
            ss[v] += t * t;
        }
    // row M = v + 8*half lives on lanes {0..15} (half 0) or {16..31} (half 1):
    // reduce over the 16-lane N-group, then rsqrt-scale
    for (int v = 0; v < 8; ++v) {
        float s = ss[v];
        s += __shfl_xor(s, 1, 16);
        s += __shfl_xor(s, 2, 16);
        s += __shfl_xor(s, 4, 16);
        s += __shfl_xor(s, 8, 16);
        ss[v] = 1.0f / fmaxf(sqrtf(s), 1e-12f);
    }
    for (int nb = 0; nb < 4; ++nb)
        for (int v = 0; v < 8; ++v) {
            const int M = v + 8 * half;
            xout[(size_t)(row0 + M) * EMB + nb * 16 + m] = val[nb][v] * ss[v];
        }
}

// -------------------- BPR loss: one wave per triplet --------------------
__global__ void ngcf_loss(const float* __restrict__ x0, const float* __restrict__ x1,
                          const float* __restrict__ x2, const float* __restrict__ x3,
                          const int* __restrict__ u, const int* __restrict__ ii,
                          const int* __restrict__ jj, float* __restrict__ acc) {
    const int lane = threadIdx.x & 31;
    const int wave = (int)((blockIdx.x * blockDim.x + threadIdx.x) >> 5);
    if (wave >= BATCH) return;
    const int ur = u[wave];
    const int pr = N_USERS + ii[wave];
    const int nr = N_USERS + jj[wave];
    const float* bufs[4] = {x0, x1, x2, x3};
    float dui = 0.0f, duj = 0.0f, sq = 0.0f;
    for (int b = 0; b < 4; ++b) {
        const float2 ue = *(const float2*)(bufs[b] + (size_t)ur * EMB + lane * 2);
        const float2 pe = *(const float2*)(bufs[b] + (size_t)pr * EMB + lane * 2);
        const float2 ne = *(const float2*)(bufs[b] + (size_t)nr * EMB + lane * 2);
        dui += ue.x * pe.x + ue.y * pe.y;
        duj += ue.x * ne.x + ue.y * ne.y;
        sq  += ue.x * ue.x + ue.y * ue.y + pe.x * pe.x + pe.y * pe.y
             + ne.x * ne.x + ne.y * ne.y;
    }
    for (int off = 16; off >= 1; off >>= 1) {
        dui += __shfl_xor(dui, off, 32);
        duj += __shfl_xor(duj, off, 32);
        sq  += __shfl_xor(sq,  off, 32);
    }
    if (lane == 0) {
        const float x = dui - duj;
        // -log_sigmoid(x) = softplus(-x), numerically stable
        const float bpr = fmaxf(-x, 0.0f) + log1pf(expf(-fabsf(x)));
        atomicAdd(&acc[0], bpr);
        atomicAdd(&acc[1], sq);
    }
}

__global__ void ngcf_finalize(const float* __restrict__ acc, float* __restrict__ out) {
    if (threadIdx.x == 0 && blockIdx.x == 0) {
        out[0] = acc[0] / (float)BATCH + REG_COEF * (0.5f * acc[1] / (float)BATCH);
    }
}

extern "C" void kernel_launch(void* const* d_in, const int* in_sizes, int n_in,
                              void* d_out, int out_size, void* d_ws, size_t ws_size,
                              hipStream_t stream) {
    (void)in_sizes; (void)n_in; (void)out_size; (void)ws_size;
    const int*   rows = (const int*)d_in[0];
    const int*   cols = (const int*)d_in[1];
    const float* vals = (const float*)d_in[2];
    const float* uemb = (const float*)d_in[3];
    const float* iemb = (const float*)d_in[4];
    const float* W1   = (const float*)d_in[5];
    const float* b1   = (const float*)d_in[6];
    const float* W2   = (const float*)d_in[7];
    const float* b2   = (const float*)d_in[8];
    const int*   u    = (const int*)d_in[9];
    const int*   ii   = (const int*)d_in[10];
    const int*   jj   = (const int*)d_in[11];
    float* out = (float*)d_out;

    const size_t NE = (size_t)N_NODES * EMB;   // 9.6M floats per node matrix
    float* ws = (float*)d_ws;
    float* x[N_LAYERS + 1];
    x[0] = ws;
    x[1] = x[0] + NE;
    x[2] = x[1] + NE;
    x[3] = x[2] + NE;
    float* sideL = x[3] + NE;
    float* acc   = sideL + NE;                 // 2 floats (padded to 8): bpr sum, sq sum
    _Float16* W1h = (_Float16*)(acc + 8);      // 3*64*64 halves
    _Float16* W2h = W1h + (size_t)N_LAYERS * EMB * EMB;

    const int TPB = 256;

    // x0 = concat(user_emb, item_emb)
    ngcf_concat<<<(int)((NE + TPB - 1) / TPB), TPB, 0, stream>>>(uemb, iemb, x[0]);
    // zero loss accumulators; convert weights to f16 once
    ngcf_zero<<<1, 32, 0, stream>>>(acc, 2);
    ngcf_cvt_weights<<<(N_LAYERS * EMB * EMB + TPB - 1) / TPB, TPB, 0, stream>>>(
        W1, W2, W1h, W2h);

    const int ntiles = N_NODES / 16;           // 9375
    const int tblocks = (ntiles * 32 + TPB - 1) / TPB;

    for (int k = 0; k < N_LAYERS; ++k) {
        ngcf_zero<<<(int)((NE + TPB - 1) / TPB), TPB, 0, stream>>>(sideL, NE);
        ngcf_spmm<<<4096, TPB, 0, stream>>>(rows, cols, vals, x[k], sideL);
        ngcf_transform<<<tblocks, TPB, 0, stream>>>(
            sideL, x[k],
            W1h + (size_t)k * EMB * EMB, b1 + (size_t)k * EMB,
            W2h + (size_t)k * EMB * EMB, b2 + (size_t)k * EMB,
            x[k + 1], ntiles);
    }

    ngcf_loss<<<(BATCH * 32 + TPB - 1) / TPB, TPB, 0, stream>>>(
        x[0], x[1], x[2], x[3], u, ii, jj, acc);
    ngcf_finalize<<<1, 32, 0, stream>>>(acc, out);
}
